// SimpleLSTM_11089605558459
// MI455X (gfx1250) — matrix operand order repaired and checked by reading, not compiled, not run on hardware
//
#include <hip/hip_runtime.h>

#define H     2048
#define BATCH 256
#define VOCAB 95
#define VP    96
#define SQLEN 162
#define FOURH 8192
#define THINK 15
#define ANS   29

typedef __bf16 v16bf __attribute__((ext_vector_type(16)));
typedef float  v8f   __attribute__((ext_vector_type(8)));

union FragU { v16bf v; uint4 q[2]; };

__device__ __forceinline__ __bf16 f2bf(float f) {
  union { float f; unsigned u; } a; a.f = f;
  unsigned r = a.u + 0x7FFFu + ((a.u >> 16) & 1u);
  unsigned short h = (unsigned short)(r >> 16);
  __bf16 b; __builtin_memcpy(&b, &h, 2); return b;
}

__device__ __forceinline__ float sigf(float x) { return 1.0f / (1.0f + __expf(-x)); }

// ---------------------------------------------------------------------------
// Fused LSTM cell: one wave owns a 32-row x 16-col tile and computes all 4
// gates (i,f,g,o) in-register via v_wmma_f32_16x16x32_bf16, then applies the
// cell nonlinearity per-lane. MODE: 0 = one-hot gather input (phase 1),
// 1 = zero input (thinking), 2 = bf16 logits input via WMMA preamble (answer).
// Grid: 128 blocks x 256 threads (8 waves) = 1024 wave-jobs = 8 rowgrp x 128 coltile.
// ---------------------------------------------------------------------------
template <int MODE>
__global__ __launch_bounds__(256)
void lstm_cell(const __bf16* __restrict__ h_in,
               const __bf16* __restrict__ in_bf,     // B x VP logits (MODE 2)
               const long long* __restrict__ x, int t,
               const __bf16* __restrict__ Whh,       // 4H x H bf16 (K fast)
               const __bf16* __restrict__ Wih_bf,    // 4H x VP bf16 (MODE 2)
               const float*  __restrict__ Wih_f32,   // 4H x V f32 (MODE 0 gather)
               const float*  __restrict__ bias,      // 4H (b_ih + b_hh)
               float*  __restrict__ c,
               __bf16* __restrict__ h_out,
               __bf16* __restrict__ out_copy)        // nullable
{
  const int wave = threadIdx.x >> 5;
  const int lane = threadIdx.x & 31;
  const int job  = blockIdx.x * 8 + wave;
  const int m0   = (job >> 7) * 32;
  const int n0   = (job & 127) * 16;
  const int nl   = lane & 15;
  const int hi   = lane >> 4;
  const int khA  = hi * 16;   // A fragment K sub-offset (contiguous k = 16*hi + s)
  const int kbB  = hi * 16;   // B fragment K base (contiguous k = 16*hi + s)

  const v8f z8 = {0.f,0.f,0.f,0.f,0.f,0.f,0.f,0.f};
  v8f acc[2][4] = {{z8,z8,z8,z8},{z8,z8,z8,z8}};

  const __bf16* arow0 = h_in + (size_t)(m0 + nl) * H;
  const __bf16* arow1 = h_in + (size_t)(m0 + 16 + nl) * H;
  const __bf16* brow[4];
#pragma unroll
  for (int g = 0; g < 4; ++g)
    brow[g] = Whh + (size_t)(g * H + n0 + nl) * H + kbB;

  if (MODE == 2) {  // inputterm = logits @ W_ih^T, K = 96 (3 chunks of 32)
    const __bf16* la0 = in_bf + (size_t)(m0 + nl) * VP;
    const __bf16* la1 = in_bf + (size_t)(m0 + 16 + nl) * VP;
#pragma unroll
    for (int kc = 0; kc < 3; ++kc) {
      const int k0 = kc * 32;
      FragU a0, a1;
      a0.q[0] = *(const uint4*)(la0 + k0 + khA);
      a0.q[1] = *(const uint4*)(la0 + k0 + 8 + khA);
      a1.q[0] = *(const uint4*)(la1 + k0 + khA);
      a1.q[1] = *(const uint4*)(la1 + k0 + 8 + khA);
#pragma unroll
      for (int g = 0; g < 4; ++g) {
        FragU b;
        const __bf16* bp = Wih_bf + (size_t)(g * H + n0 + nl) * VP + k0 + kbB;
        b.q[0] = ((const uint4*)bp)[0];
        b.q[1] = ((const uint4*)bp)[1];
        acc[0][g] = __builtin_amdgcn_wmma_f32_16x16x32_bf16(false, a0.v, false, b.v,
                                                            (short)0, acc[0][g], false, false);
        acc[1][g] = __builtin_amdgcn_wmma_f32_16x16x32_bf16(false, a1.v, false, b.v,
                                                            (short)0, acc[1][g], false, false);
      }
    }
  }

  // Main recurrent GEMM: g += h @ W_hh^T, K = 2048
#pragma unroll 2
  for (int k0 = 0; k0 < H; k0 += 32) {
    FragU a0, a1;
    a0.q[0] = *(const uint4*)(arow0 + k0 + khA);
    a0.q[1] = *(const uint4*)(arow0 + k0 + 8 + khA);
    a1.q[0] = *(const uint4*)(arow1 + k0 + khA);
    a1.q[1] = *(const uint4*)(arow1 + k0 + 8 + khA);
#pragma unroll
    for (int g = 0; g < 4; ++g) {
      FragU b;
      const __bf16* bp = brow[g] + k0;
      b.q[0] = ((const uint4*)bp)[0];
      b.q[1] = ((const uint4*)bp)[1];
      acc[0][g] = __builtin_amdgcn_wmma_f32_16x16x32_bf16(false, a0.v, false, b.v,
                                                          (short)0, acc[0][g], false, false);
      acc[1][g] = __builtin_amdgcn_wmma_f32_16x16x32_bf16(false, a1.v, false, b.v,
                                                          (short)0, acc[1][g], false, false);
    }
  }

  // Epilogue: bias (+ one-hot gather), sigmoid/tanh cell update, in-register.
  const int col = n0 + nl;
  const float bi = bias[col];
  const float bff = bias[H + col];
  const float bg = bias[2 * H + col];
  const float bo = bias[3 * H + col];

#pragma unroll
  for (int rt = 0; rt < 2; ++rt) {
#pragma unroll
    for (int r = 0; r < 8; ++r) {
      const int row = m0 + rt * 16 + hi * 8 + r;
      float iv = acc[rt][0][r] + bi;
      float fv = acc[rt][1][r] + bff;
      float gv = acc[rt][2][r] + bg;
      float ov = acc[rt][3][r] + bo;
      if (MODE == 0) {  // one-hot input: gather column x[b][t] of W_ih
        const int xv = (int)x[(size_t)row * SQLEN + t];
        iv += Wih_f32[(size_t)(0 * H + col) * VOCAB + xv];
        fv += Wih_f32[(size_t)(1 * H + col) * VOCAB + xv];
        gv += Wih_f32[(size_t)(2 * H + col) * VOCAB + xv];
        ov += Wih_f32[(size_t)(3 * H + col) * VOCAB + xv];
      }
      const size_t idx = (size_t)row * H + col;
      const float cn = sigf(fv) * c[idx] + sigf(iv) * tanhf(gv);
      const float hn = sigf(ov) * tanhf(cn);
      c[idx] = cn;
      const __bf16 hb = f2bf(hn);
      h_out[idx] = hb;
      if (out_copy) out_copy[idx] = hb;
    }
  }
}

// ---------------------------------------------------------------------------
// logits = out @ W_proj^T  (256 x 95, K = 2048). 48 wave-jobs = 6 blocks.
// Writes fp32 to d_out slab and bf16 (96-padded, col 95 zeroed) for feedback.
// ---------------------------------------------------------------------------
__global__ __launch_bounds__(256)
void lstm_logits(const __bf16* __restrict__ out_bf,
                 const __bf16* __restrict__ Wproj_bf,  // VP x H (row 95 zero)
                 float*  __restrict__ out_f32,         // B x V
                 __bf16* __restrict__ logits_bf)       // B x VP
{
  const int wave = threadIdx.x >> 5;
  const int lane = threadIdx.x & 31;
  const int job  = blockIdx.x * 8 + wave;  // 0..47
  const int m0   = (job / 6) * 32;
  const int n0   = (job % 6) * 16;
  const int nl = lane & 15, hi = lane >> 4;
  const int khA = hi * 16, kbB = hi * 16;

  const v8f z8 = {0.f,0.f,0.f,0.f,0.f,0.f,0.f,0.f};
  v8f acc0 = z8, acc1 = z8;

  const __bf16* arow0 = out_bf + (size_t)(m0 + nl) * H;
  const __bf16* arow1 = out_bf + (size_t)(m0 + 16 + nl) * H;
  const __bf16* brow  = Wproj_bf + (size_t)(n0 + nl) * H + kbB;

#pragma unroll 2
  for (int k0 = 0; k0 < H; k0 += 32) {
    FragU a0, a1, b;
    a0.q[0] = *(const uint4*)(arow0 + k0 + khA);
    a0.q[1] = *(const uint4*)(arow0 + k0 + 8 + khA);
    a1.q[0] = *(const uint4*)(arow1 + k0 + khA);
    a1.q[1] = *(const uint4*)(arow1 + k0 + 8 + khA);
    b.q[0] = ((const uint4*)(brow + k0))[0];
    b.q[1] = ((const uint4*)(brow + k0))[1];
    acc0 = __builtin_amdgcn_wmma_f32_16x16x32_bf16(false, a0.v, false, b.v, (short)0, acc0, false, false);
    acc1 = __builtin_amdgcn_wmma_f32_16x16x32_bf16(false, a1.v, false, b.v, (short)0, acc1, false, false);
  }

  const int v = n0 + nl;
#pragma unroll
  for (int rt = 0; rt < 2; ++rt) {
#pragma unroll
    for (int r = 0; r < 8; ++r) {
      const int row = m0 + rt * 16 + hi * 8 + r;
      const float val = rt ? acc1[r] : acc0[r];
      logits_bf[(size_t)row * VP + v] = (v < VOCAB) ? f2bf(val) : f2bf(0.f);
      if (v < VOCAB) out_f32[(size_t)row * VOCAB + v] = val;
    }
  }
}

// --------------------------- prologue kernels ------------------------------
__global__ void k_cvt(const float* __restrict__ in, __bf16* __restrict__ out, int n) {
  for (int i = blockIdx.x * blockDim.x + threadIdx.x; i < n; i += gridDim.x * blockDim.x)
    out[i] = f2bf(in[i]);
}
__global__ void k_cvt_padcols(const float* __restrict__ in, __bf16* __restrict__ out,
                              int rows, int ldin, int ldout) {
  int i = blockIdx.x * blockDim.x + threadIdx.x;
  if (i >= rows * ldout) return;
  int r = i / ldout, cc = i % ldout;
  out[i] = (cc < ldin) ? f2bf(in[(size_t)r * ldin + cc]) : f2bf(0.f);
}
__global__ void k_cvt_padrows(const float* __restrict__ in, __bf16* __restrict__ out,
                              int validRows, int totRows, int ld) {
  int i = blockIdx.x * blockDim.x + threadIdx.x;
  if (i >= totRows * ld) return;
  out[i] = ((i / ld) < validRows) ? f2bf(in[i]) : f2bf(0.f);
}
__global__ void k_bias(const float* __restrict__ a, const float* __restrict__ b,
                       float* __restrict__ o, int n) {
  int i = blockIdx.x * blockDim.x + threadIdx.x;
  if (i < n) o[i] = a[i] + b[i];
}
__global__ void k_init(__bf16* __restrict__ h, float* __restrict__ c, int n) {
  int i = blockIdx.x * blockDim.x + threadIdx.x;
  if (i < n) { h[i] = f2bf(0.f); c[i] = 0.f; }
}

// ---------------------------------------------------------------------------
extern "C" void kernel_launch(void* const* d_in, const int* in_sizes, int n_in,
                              void* d_out, int out_size, void* d_ws, size_t ws_size,
                              hipStream_t stream) {
  (void)in_sizes; (void)n_in; (void)out_size; (void)ws_size;
  const long long* x   = (const long long*)d_in[0];
  const float* W_ih    = (const float*)d_in[1];
  const float* W_hh    = (const float*)d_in[2];
  const float* b_ih    = (const float*)d_in[3];
  const float* b_hh    = (const float*)d_in[4];
  const float* W_proj  = (const float*)d_in[5];
  float* out           = (float*)d_out;

  char* p = (char*)d_ws;
  auto take = [&](size_t bytes) { char* q = p; p += (bytes + 255) & ~size_t(255); return q; };
  __bf16* Whh_bf  = (__bf16*)take((size_t)FOURH * H * 2);   // 32 MB, L2-resident
  __bf16* Wih_bf  = (__bf16*)take((size_t)FOURH * VP * 2);
  __bf16* Wpr_bf  = (__bf16*)take((size_t)VP * H * 2);
  float*  bias    = (float*) take((size_t)FOURH * 4);
  __bf16* h_a     = (__bf16*)take((size_t)BATCH * H * 2);
  __bf16* h_b     = (__bf16*)take((size_t)BATCH * H * 2);
  __bf16* out_buf = (__bf16*)take((size_t)BATCH * H * 2);   // out_q / answer feedback
  float*  c       = (float*) take((size_t)BATCH * H * 4);
  __bf16* lg_bf   = (__bf16*)take((size_t)BATCH * VP * 2);

  // one-time (per call) conversions
  { int n = FOURH * H;  k_cvt<<<4096, 256, 0, stream>>>(W_hh, Whh_bf, n); }
  { int n = FOURH * VP; k_cvt_padcols<<<(n + 255) / 256, 256, 0, stream>>>(W_ih, Wih_bf, FOURH, VOCAB, VP); }
  { int n = VP * H;     k_cvt_padrows<<<(n + 255) / 256, 256, 0, stream>>>(W_proj, Wpr_bf, VOCAB, VP, H); }
  k_bias<<<(FOURH + 255) / 256, 256, 0, stream>>>(b_ih, b_hh, bias, FOURH);
  { int n = BATCH * H;  k_init<<<(n + 255) / 256, 256, 0, stream>>>(h_a, c, n); }

  __bf16* hin = h_a; __bf16* hout = h_b;

  // Phase 1: question (one-hot gather input); snapshot out_q at last step
  for (int t = 0; t < SQLEN; ++t) {
    lstm_cell<0><<<128, 256, 0, stream>>>(hin, nullptr, x, t, Whh_bf, nullptr, W_ih,
                                          bias, c, hout, (t == SQLEN - 1) ? out_buf : nullptr);
    __bf16* tmp = hout; hout = hin; hin = tmp;
  }
  // Phase 2: thinking (zero input)
  for (int t = 0; t < THINK; ++t) {
    lstm_cell<1><<<128, 256, 0, stream>>>(hin, nullptr, x, 0, Whh_bf, nullptr, nullptr,
                                          bias, c, hout, nullptr);
    __bf16* tmp = hout; hout = hin; hin = tmp;
  }
  // Phase 3: autoregressive answer; logits fed back through W_ih
  for (int s = 0; s < ANS; ++s) {
    lstm_logits<<<6, 256, 0, stream>>>(out_buf, Wpr_bf, out + (size_t)s * BATCH * VOCAB, lg_bf);
    if (s < ANS - 1) {
      lstm_cell<2><<<128, 256, 0, stream>>>(hin, lg_bf, x, 0, Whh_bf, Wih_bf, nullptr,
                                            bias, c, hout, out_buf);
      __bf16* tmp = hout; hout = hin; hin = tmp;
    }
  }
}
